// CompoundMultivariateEmbedding_9380208574576
// MI455X (gfx1250) — compile-verified, matrix operand order checked by reading, not run on hardware
//
#include <hip/hip_runtime.h>

// ---- problem constants (match reference) ----
#define EMBED    128
#define ATTR     25
#define REM      28
#define N_LEVELS 50
#define N_TYPES  2
#define N_FEATS  2
#define N_EXCH   3
#define N_PAIRS  20

#define ROWS     64          // rows per tile
#define THREADS  128         // 4 waves (wave32)
#define XSTRIDE  68          // dword stride of a 128-col bf16-pair row (padded, 16B aligned)
#define MAX_GRID 2048        // blocks; each grid-strides over tiles, amortizing W staging

// table offsets inside flat LDS table buffer (in floats)
#define OFF_L 0
#define OFF_T (N_LEVELS*ATTR)                    // 1250
#define OFF_F (OFF_T + N_TYPES*ATTR)             // 1300
#define OFF_E (OFF_F + N_FEATS*ATTR)             // 1350
#define OFF_P (OFF_E + N_EXCH*ATTR)              // 1425
#define TAB_TOT (OFF_P + N_PAIRS*REM)            // 1985

typedef __bf16 bf16x16 __attribute__((ext_vector_type(16)));
typedef float  f32x8   __attribute__((ext_vector_type(8)));

union Frag { unsigned int u[8]; bf16x16 v; };

// round-to-nearest-even f32 -> bf16, pack two columns into one dword
__device__ __forceinline__ unsigned int packbf(float lo, float hi) {
    unsigned int a = __float_as_uint(lo);
    unsigned int b = __float_as_uint(hi);
    a += 0x7FFFu + ((a >> 16) & 1u);
    b += 0x7FFFu + ((b >> 16) & 1u);
    return (a >> 16) | (b & 0xFFFF0000u);
}

// gather one concatenated column; with compile-time c this folds to one LDS load
__device__ __forceinline__ float gcol(int c, const float* l, const float* t,
                                      const float* f, const float* e, const float* p) {
    if (c < 25)       return l[c];
    else if (c < 50)  return t[c - 25];
    else if (c < 75)  return f[c - 50];
    else if (c < 100) return e[c - 75];
    else              return p[c - 100];
}

// build 64 columns [CB, CB+64) of one activation row as packed bf16 pairs.
// CB is a template parameter so every gcol() branch resolves at compile time.
template<int CB>
__device__ __forceinline__ void buildHalf(unsigned int* __restrict__ xrow,
                                          const float* l, const float* t,
                                          const float* f, const float* e,
                                          const float* p) {
    #pragma unroll
    for (int c2 = 0; c2 < 32; ++c2) {
        float f0 = gcol(CB + 2*c2,     l, t, f, e, p);
        float f1 = gcol(CB + 2*c2 + 1, l, t, f, e, p);
        xrow[c2] = packbf(f0, f1);
    }
}

// -------- main kernel: grid-stride over FULL 64-row tiles (no guards) --------
__global__ void __launch_bounds__(THREADS)
embed_wmma_kernel(const float* __restrict__ level_tab,
                  const float* __restrict__ type_tab,
                  const float* __restrict__ feat_tab,
                  const float* __restrict__ exch_tab,
                  const float* __restrict__ pair_tab,
                  const float* __restrict__ W,
                  const float* __restrict__ bias,
                  const int*   __restrict__ level_idx,
                  const int*   __restrict__ type_idx,
                  const int*   __restrict__ feat_idx,
                  const int*   __restrict__ exch_idx,
                  const int*   __restrict__ pair_idx,
                  float*       __restrict__ out,
                  int numTiles)
{
    __shared__ __align__(16) unsigned int Xs32[ROWS  * XSTRIDE];   // 17,408 B bf16 activation tile
    __shared__ __align__(16) unsigned int Ws32[EMBED * XSTRIDE];   // 34,816 B bf16 weight matrix
    __shared__ float tabS[TAB_TOT];                                //  7,940 B tables
    __shared__ float biasS[EMBED];                                 //    512 B

    const int tid = threadIdx.x;

    // ---------------- stage tables, bias, W (fp32 -> bf16) ONCE per block --------
    for (int i = tid; i < N_LEVELS*ATTR; i += THREADS) tabS[OFF_L + i] = level_tab[i];
    for (int i = tid; i < N_TYPES *ATTR; i += THREADS) tabS[OFF_T + i] = type_tab[i];
    for (int i = tid; i < N_FEATS *ATTR; i += THREADS) tabS[OFF_F + i] = feat_tab[i];
    for (int i = tid; i < N_EXCH  *ATTR; i += THREADS) tabS[OFF_E + i] = exch_tab[i];
    for (int i = tid; i < N_PAIRS *REM;  i += THREADS) tabS[OFF_P + i] = pair_tab[i];
    for (int i = tid; i < EMBED;         i += THREADS) biasS[i] = bias[i];

    const float4* W4 = (const float4*)W;              // 128x128 fp32 = 4096 float4
    for (int i = tid; i < EMBED * (EMBED/4); i += THREADS) {
        float4 w = W4[i];
        int e  = i >> 5;          // 32 float4 per row
        int f4 = i & 31;
        Ws32[e * XSTRIDE + 2*f4    ] = packbf(w.x, w.y);
        Ws32[e * XSTRIDE + 2*f4 + 1] = packbf(w.z, w.w);
    }
    __syncthreads();

    // loop-invariant lane decomposition
    const int r64  = tid & 63;          // build: row within tile
    const int wave = tid >> 5;          // 0..3, owns rows [16*wave, 16*wave+16)
    const int lane = tid & 31;
    const int m16  = lane & 15;
    const int half = lane >> 4;
    const int khB  = 4 * half;
    const int arow = (wave * 16 + m16) * XSTRIDE;

    for (int tile = blockIdx.x; tile < numTiles; tile += gridDim.x) {
        const int rowBase = tile * ROWS;

        // ---------------- build activation tile (gather + concat -> bf16) ------
        {
            const int nr = rowBase + r64;             // always in range (full tiles only)
            const float* lrow = tabS + OFF_L + level_idx[nr] * ATTR;
            const float* trow = tabS + OFF_T + type_idx[nr]  * ATTR;
            const float* frow = tabS + OFF_F + feat_idx[nr]  * ATTR;
            const float* erow = tabS + OFF_E + exch_idx[nr]  * ATTR;
            const float* prow = tabS + OFF_P + pair_idx[nr]  * REM;
            if (tid < 64) {                            // wave-uniform branch
                buildHalf<0 >(Xs32 + r64 * XSTRIDE,      lrow, trow, frow, erow, prow);
            } else {
                buildHalf<64>(Xs32 + r64 * XSTRIDE + 32, lrow, trow, frow, erow, prow);
            }
        }
        __syncthreads();

        // ---------------- WMMA: y[16x128] per wave = x[16x128] * W^T + b -------
        // A fragments: 16-bit A 16x32 layout; dword i holds K-pair (i<4 ? i : 4+i) + 4*half
        Frag a[4];
        #pragma unroll
        for (int s = 0; s < 4; ++s) {
            #pragma unroll
            for (int i = 0; i < 8; ++i) {
                int kh = ((i < 4) ? i : (4 + i)) + khB + 16 * s;
                a[s].u[i] = Xs32[arow + kh];
            }
        }

        // single per-lane output base; all 64 stores use compile-time immediate offsets
        float* const outp = out + (size_t)(rowBase + wave * 16 + 8 * half) * EMBED + m16;

        #pragma unroll
        for (int t = 0; t < 8; ++t) {                 // 8 tiles of 16 output columns
            f32x8 acc = {0.f, 0.f, 0.f, 0.f, 0.f, 0.f, 0.f, 0.f};
            const int brow = (t * 16 + m16) * XSTRIDE;   // B[k][n] = W[16t+n][k]
            #pragma unroll
            for (int s = 0; s < 4; ++s) {             // K = 128 in 4 steps of 32
                Frag bf;
                #pragma unroll
                for (int i = 0; i < 8; ++i) {
                    int kh = ((i < 4) ? i : (4 + i)) + khB + 16 * s;
                    bf.u[i] = Ws32[brow + kh];
                }
                acc = __builtin_amdgcn_wmma_f32_16x16x32_bf16(
                          false, a[s].v, false, bf.v, (short)0, acc, false, false);
            }
            // C/D layout: col = lane%16, row = vgpr + 8*half
            const float bv = biasS[t * 16 + m16];
            #pragma unroll
            for (int r8 = 0; r8 < 8; ++r8)           // streaming output: NT stores
                __builtin_nontemporal_store(acc[r8] + bv,
                                            outp + (size_t)(r8 * EMBED + t * 16));
        }
        __syncthreads();    // Xs32 rewritten next iteration
    }
}

// -------- tail kernel: scalar fp32, handles rows [start, n) (n % 64 rows) --------
__global__ void embed_tail_kernel(const float* __restrict__ level_tab,
                                  const float* __restrict__ type_tab,
                                  const float* __restrict__ feat_tab,
                                  const float* __restrict__ exch_tab,
                                  const float* __restrict__ pair_tab,
                                  const float* __restrict__ W,
                                  const float* __restrict__ bias,
                                  const int*   __restrict__ level_idx,
                                  const int*   __restrict__ type_idx,
                                  const int*   __restrict__ feat_idx,
                                  const int*   __restrict__ exch_idx,
                                  const int*   __restrict__ pair_idx,
                                  float*       __restrict__ out,
                                  int start, int n)
{
    const int gid = blockIdx.x * blockDim.x + threadIdx.x;
    const int row = start + (gid >> 7);
    const int e   = gid & 127;
    if (row >= n) return;
    const float* lrow = level_tab + level_idx[row] * ATTR;
    const float* trow = type_tab  + type_idx[row]  * ATTR;
    const float* frow = feat_tab  + feat_idx[row]  * ATTR;
    const float* erow = exch_tab  + exch_idx[row]  * ATTR;
    const float* prow = pair_tab  + pair_idx[row]  * REM;
    const float* wrow = W + e * EMBED;
    float acc = bias[e];
    for (int c = 0; c < EMBED; ++c)
        acc += gcol(c, lrow, trow, frow, erow, prow) * wrow[c];
    out[(size_t)row * EMBED + e] = acc;
}

extern "C" void kernel_launch(void* const* d_in, const int* in_sizes, int n_in,
                              void* d_out, int out_size, void* d_ws, size_t ws_size,
                              hipStream_t stream) {
    const float* level_tab = (const float*)d_in[0];
    const float* type_tab  = (const float*)d_in[1];
    const float* feat_tab  = (const float*)d_in[2];
    const float* exch_tab  = (const float*)d_in[3];
    const float* pair_tab  = (const float*)d_in[4];
    const float* W         = (const float*)d_in[5];
    const float* b         = (const float*)d_in[6];
    const int* li = (const int*)d_in[7];
    const int* ti = (const int*)d_in[8];
    const int* fi = (const int*)d_in[9];
    const int* ei = (const int*)d_in[10];
    const int* pi = (const int*)d_in[11];
    const int  n  = in_sizes[7];

    const int numTiles = n / ROWS;              // 16384 for N = 2^20
    if (numTiles > 0) {
        const int grid = numTiles < MAX_GRID ? numTiles : MAX_GRID;
        embed_wmma_kernel<<<grid, THREADS, 0, stream>>>(
            level_tab, type_tab, feat_tab, exch_tab, pair_tab, W, b,
            li, ti, fi, ei, pi, (float*)d_out, numTiles);
    }
    const int tailStart = numTiles * ROWS;
    const int tailRows  = n - tailStart;        // 0 for N = 2^20
    if (tailRows > 0) {
        const int tailThreads = tailRows * EMBED;
        embed_tail_kernel<<<(tailThreads + 127) / 128, 128, 0, stream>>>(
            level_tab, type_tab, feat_tab, exch_tab, pair_tab, W, b,
            li, ti, fi, ei, pi, (float*)d_out, tailStart, n);
    }
}
